// CCEncoderVarDepth_23725399343723
// MI455X (gfx1250) — matrix operand order repaired and checked by reading, not compiled, not run on hardware
//
#include <hip/hip_runtime.h>
#include <hip/hip_bf16.h>
#include <math.h>

// ---------------------------------------------------------------------------
// CC encoder (18x conv3x3 + SiLU + BN + avgpool) as bf16 implicit-GEMM WMMA
// with TDM (tensor_load_to_lds) weight staging.
// Activations: bf16 NHWC in workspace. Weights are pre-transposed at convert
// time to wT[Cout][Ktot] (Ktot = 9*Cin) so the TDM DMA lands directly in the
// Bt[n][k] fragment layout in LDS (no LDS transpose pass).
// Layer 0 is padded to Cin=4 so all layers share one vectorized path.
// Input flatten assumption (setup_inputs dict order):
//   d_in[0]=x f32 [32,1,768,256]; d_in[1..18]=conv w; d_in[19..30]=gamma;
//   d_in[31..42]=beta; d_in[43]=batch_size (unused).
// d_out = 2 * 32 * 512 f32 (head duplicated).
// ---------------------------------------------------------------------------

typedef __attribute__((ext_vector_type(16))) __bf16 v16bf;
typedef __attribute__((ext_vector_type(8)))  __bf16 v8bf;
typedef __attribute__((ext_vector_type(4)))  __bf16 v4bf;
typedef __attribute__((ext_vector_type(8)))  float  v8f;
typedef __attribute__((ext_vector_type(4)))  unsigned int u32x4;
typedef __attribute__((ext_vector_type(8)))  unsigned int u32x8;

#define BM 256   // block M tile (8 waves x 32 pixels)
#define BN 64    // block N tile (4 WMMA subtiles)
#define TK 32    // WMMA K per step (bf16)

// w f32 [Ktot][Cout] -> bf16 transposed [Cout][Ktot]
__global__ __launch_bounds__(256)
void cc_cvt_wT(const float* __restrict__ src, __bf16* __restrict__ dst,
               int Ktot, int Cout, long total) {
  long stridetot = (long)gridDim.x * blockDim.x;
  for (long i = (long)blockIdx.x * blockDim.x + threadIdx.x; i < total; i += stridetot) {
    int n = (int)(i / Ktot);
    int k = (int)(i - (long)n * Ktot);
    dst[i] = (__bf16)src[(long)k * Cout + n];
  }
}

// w0 f32 [9][1][16] -> bf16 transposed+padded [16][36] (ci 1..3 zero)
__global__ __launch_bounds__(256)
void cc_cvt_w0T_pad4(const float* __restrict__ src, __bf16* __restrict__ dst) {
  int i = blockIdx.x * 256 + threadIdx.x;
  if (i < 16 * 36) {
    int n = i / 36, k = i - n * 36;
    int ci = k & 3, p = k >> 2;
    dst[i] = (ci == 0) ? (__bf16)src[p * 16 + n] : (__bf16)0.0f;
  }
}

// x f32 [npix] (C=1) -> bf16 NHWC [npix][4], channels 1..3 zero.
__global__ __launch_bounds__(256)
void cc_cvt_in_pad4(const float* __restrict__ src, __bf16* __restrict__ dst, long npix) {
  long stridetot = (long)gridDim.x * blockDim.x;
  for (long i = (long)blockIdx.x * blockDim.x + threadIdx.x; i < npix; i += stridetot) {
    v4bf v = {};
    v[0] = (__bf16)src[i];
    *(v4bf*)&dst[i * 4] = v;
  }
}

__global__ __launch_bounds__(256)
void cc_zero_f32(float* __restrict__ p, int n) {
  int i = blockIdx.x * 256 + threadIdx.x;
  if (i < n) p[i] = 0.0f;
}

// Implicit-GEMM 3x3 conv, pad=1, NHWC bf16, f32 accumulate via WMMA.
// Wave computes 32(M) x 64(N) = 8 WMMA tiles; K = 9*Cin in chunks of 32.
// Weight panel DMA'd by the Tensor Data Mover straight into Bt[n][k].
__global__ __launch_bounds__(256)
void cc_conv3x3_wmma_bf16(const __bf16* __restrict__ in, const __bf16* __restrict__ wgtT,
                          __bf16* __restrict__ out,
                          int Hin, int Win, int Cin, int cinShift,
                          int Hout, int Wout, int woShift, int Cout,
                          int stride, int applySilu, long Mtot)
{
  __shared__ __bf16 As[BM][TK];   // im2col A tile, 16 KB
  __shared__ __bf16 Bt[BN][TK];   // TDM destination = [n][k] fragment layout, 4 KB

  const int tid  = threadIdx.x;
  const int wave = tid >> 5;
  const int lane = tid & 31;
  const int half = lane >> 4;
  const int l16  = lane & 15;

  const long mBase = (long)blockIdx.x * BM;
  const int  nBase = blockIdx.y * BN;
  const int  Ktot  = 9 << cinShift;           // 9*Cin (Cin power of two >= 4)
  const long HW    = (long)Hout * Wout;
  const int  cinMask = Cin - 1;

  // Hoist output-pixel coordinates for this thread's 8 staging rows
  // (row = j*32 + tid>>3, channel quad q = tid&7).
  const int q    = tid & 7;
  const int rsub = tid >> 3;
  int pn4[8], piy4[8], pix4[8];
#pragma unroll
  for (int j = 0; j < 8; ++j) {
    long m  = mBase + j * 32 + rsub;
    bool v  = (m < Mtot);
    long mm = v ? m : 0;
    int  n  = (int)(mm / HW);
    int  rem = (int)(mm - (long)n * HW);
    pn4[j]  = v ? n : -1;
    piy4[j] = (rem >> woShift) * stride - 1;  // pad = 1
    pix4[j] = (rem & (Wout - 1)) * stride - 1;
  }

  v8f acc[2][4] = {};
  const unsigned int ldsBt = (unsigned int)(size_t)&Bt[0][0];  // LDS byte offset

  for (int kc = 0; kc < Ktot; kc += TK) {
    // ---- TDM: async DMA of weight panel (k-contig 32 x 64 n-rows) into Bt ----
    if (wave == 0) {
      const unsigned long long gaddr =
          (unsigned long long)(size_t)wgtT +
          (((unsigned long long)(unsigned int)nBase * (unsigned int)Ktot +
            (unsigned int)kc) << 1);
      const unsigned int td0 = (unsigned int)(Ktot - kc);     // remaining K (OOB -> 0)
      const unsigned int td1 = (unsigned int)(Cout - nBase);  // remaining N (OOB -> 0)
      u32x4 g0;
      g0[0] = 1u;                                             // count=1 (valid D#)
      g0[1] = ldsBt;                                          // lds_addr
      g0[2] = (unsigned int)gaddr;                            // global_addr[31:0]
      g0[3] = (unsigned int)((gaddr >> 32) & 0x1FFFFFFu) | (2u << 30); // msbs | type=2
      u32x8 g1;
      g1[0] = (1u << 16);                                     // data_size=1 (2 bytes)
      g1[1] = (td0 & 0xFFFFu) << 16;                          // tensor_dim0[15:0]
      g1[2] = (td0 >> 16) | ((td1 & 0xFFFFu) << 16);          // dim0 hi | tensor_dim1 lo
      g1[3] = (td1 >> 16) | ((unsigned int)TK << 16);         // dim1 hi | tile_dim0=32
      g1[4] = (unsigned int)BN;                               // tile_dim1=64
      g1[5] = (unsigned int)Ktot;                             // tensor_dim0_stride lo
      g1[6] = 0u;
      g1[7] = 0u;
      asm volatile("tensor_load_to_lds %0, %1" :: "s"(g0), "s"(g1) : "memory");
    }
    if (kc + TK < Ktot)  // warm GL2 for next K slice (global_prefetch_b8)
      __builtin_prefetch((const void*)&wgtT[(long)(nBase + (tid & 63)) * Ktot + kc + TK], 0, 1);

    // ---- stage A tile (256x32): vectorized im2col (4 channels/ldg) + fused SiLU ----
    {
      const int k0 = kc + q * 4;                 // quad never crosses (ky,kx): Cin%4==0
      const bool kv = (k0 < Ktot);
      const int ci0 = k0 & cinMask;
      const int p   = k0 >> cinShift;            // 0..8 when valid
      const int ky  = p / 3;
      const int kx  = p - ky * 3;
#pragma unroll
      for (int j = 0; j < 8; ++j) {
        v4bf v = {};
        const int iy = piy4[j] + ky;
        const int ix = pix4[j] + kx;
        if (kv && pn4[j] >= 0 && iy >= 0 && iy < Hin && ix >= 0 && ix < Win)
          v = *(const v4bf*)&in[(((long)pn4[j] * Hin + iy) * Win + ix) * (long)Cin + ci0];
        if (applySilu) {
#pragma unroll
          for (int e = 0; e < 4; ++e) {
            float f = (float)v[e];
            v[e] = (__bf16)(f / (1.0f + __expf(-f)));
          }
        }
        *(v4bf*)&As[j * 32 + rsub][q * 4] = v;
      }
    }
    if (wave == 0) __builtin_amdgcn_s_wait_tensorcnt(0);
    __syncthreads();   // As staged, Bt landed

    // ---- fragments: contiguous k order, shared by A and B ----
    // A (16-bit 16x32): lane m=l16; element j -> K = half*16 + j
    v16bf aF[2];
#pragma unroll
    for (int ms = 0; ms < 2; ++ms) {
      const __bf16* arow = &As[wave * 32 + ms * 16 + l16][0];
      v8bf a0 = *(const v8bf*)(arow + half * 16);
      v8bf a1 = *(const v8bf*)(arow + half * 16 + 8);
#pragma unroll
      for (int j = 0; j < 8; ++j) { aF[ms][j] = a0[j]; aF[ms][j + 8] = a1[j]; }
    }
    // B (16-bit 32x16): lane n=l16; element j -> K = half*16 + j (contiguous in Bt[n][.])
#pragma unroll
    for (int sub = 0; sub < 4; ++sub) {
      v16bf b = *(const v16bf*)(&Bt[sub * 16 + l16][half * 16]);
#pragma unroll
      for (int ms = 0; ms < 2; ++ms)
        acc[ms][sub] = __builtin_amdgcn_wmma_f32_16x16x32_bf16(false, aF[ms], false, b,
                                                               (short)0, acc[ms][sub],
                                                               false, false);
    }
    __syncthreads();   // reads done before next chunk overwrites As/Bt
  }

  // ---- epilogue: C/D layout -> NHWC bf16 store (lanes span contiguous cout) ----
#pragma unroll
  for (int ms = 0; ms < 2; ++ms) {
#pragma unroll
    for (int sub = 0; sub < 4; ++sub) {
      if (nBase + sub * 16 < Cout) {
#pragma unroll
        for (int r = 0; r < 8; ++r) {
          const long m = mBase + wave * 32 + ms * 16 + half * 8 + r;
          if (m < Mtot)
            out[m * (long)Cout + nBase + sub * 16 + l16] = (__bf16)acc[ms][sub][r];
        }
      }
    }
  }
}

// BN pass 1: per-channel sum/sumsq. Grid stride is a multiple of 512 (host
// forces even grid), so each thread's channel is loop-invariant -> register
// accumulation, then 2 ds_add_f32 per thread + global f32 atomics per block.
__global__ __launch_bounds__(256)
void cc_bn_reduce(const __bf16* __restrict__ x, float* __restrict__ stats,
                  long total, int C, int cMask) {
  __shared__ float s[1024];  // C <= 512: [0..C) sum, [C..2C) sumsq
  const int tid = threadIdx.x;
  for (int i = tid; i < 2 * C; i += 256) s[i] = 0.0f;
  __syncthreads();
  const long stridetot = (long)gridDim.x * 256;   // multiple of 512
  long i = (long)blockIdx.x * 256 + tid;
  const int c = (int)(i & cMask);                 // invariant: C divides stride
  float sv = 0.0f, qv = 0.0f;
  for (; i < total; i += stridetot) {
    const float v = (float)x[i];
    sv += v; qv += v * v;
  }
  atomicAdd(&s[c], sv);
  atomicAdd(&s[C + c], qv);
  __syncthreads();
  for (int k = tid; k < 2 * C; k += 256) atomicAdd(&stats[k], s[k]);
}

// BN pass 2: normalize in place
__global__ __launch_bounds__(256)
void cc_bn_apply(__bf16* __restrict__ x, const float* __restrict__ stats,
                 const float* __restrict__ gamma, const float* __restrict__ beta,
                 long total, int C, int cMask, float invCnt) {
  const long stridetot = (long)gridDim.x * 256;
  for (long i = (long)blockIdx.x * 256 + threadIdx.x; i < total; i += stridetot) {
    const int c = (int)(i & cMask);
    const float mean = stats[c] * invCnt;
    const float var  = stats[C + c] * invCnt - mean * mean;
    const float sc   = gamma[c] * rsqrtf(var + 1e-5f);
    x[i] = (__bf16)(((float)x[i] - mean) * sc + beta[c]);
  }
}

// Global average pool over H*W, f32 output, written twice (tuple of two heads).
__global__ __launch_bounds__(256)
void cc_avgpool_head(const __bf16* __restrict__ x, float* __restrict__ out,
                     int P, int C, float inv, long dupOff) {
  const int n = blockIdx.x;
  const int c = blockIdx.y * 256 + threadIdx.x;
  if (c >= C) return;
  float s = 0.0f;
  for (int p = 0; p < P; ++p) s += (float)x[((long)n * P + p) * C + c];
  const float v = s * inv;
  out[(long)n * C + c] = v;
  out[dupOff + (long)n * C + c] = v;
}

extern "C" void kernel_launch(void* const* d_in, const int* in_sizes, int n_in,
                              void* d_out, int out_size, void* d_ws, size_t ws_size,
                              hipStream_t stream)
{
  (void)in_sizes; (void)n_in; (void)out_size; (void)ws_size;
  static const int CH[6] = {16, 32, 64, 128, 256, 512};
  int ci[18], co[18], st[18];
  ci[0] = 4;     co[0] = CH[0]; st[0] = 2;   // layer 0 padded 1 -> 4 input channels
  ci[1] = CH[0]; co[1] = CH[0]; st[1] = 1;
  ci[2] = CH[0]; co[2] = CH[0]; st[2] = 1;
  int li = 3;
  for (int s = 1; s < 6; ++s) {
    ci[li] = CH[s - 1]; co[li] = CH[s]; st[li] = 2; ++li;
    ci[li] = CH[s];     co[li] = CH[s]; st[li] = 1; ++li;
    ci[li] = CH[s];     co[li] = CH[s]; st[li] = 1; ++li;
  }
  const int N = 32;

  // ---- workspace layout: bf16 weights(T) | stats | ping | pong ----
  char* ws = (char*)d_ws;
  size_t welems[18], woff[18], wtot = 0;
  for (int i = 0; i < 18; ++i) { welems[i] = (size_t)9 * ci[i] * co[i]; woff[i] = wtot; wtot += welems[i]; }
  __bf16* wbf = (__bf16*)ws;
  size_t off = ((wtot * 2 + 255) / 256) * 256;
  float* stats = (float*)(ws + off);
  off += 4096;
  long maxE = (long)N * 768 * 256 * 4;                  // padded input
  { int H = 768, W = 256;
    for (int i = 0; i < 18; ++i) { if (st[i] == 2) { H /= 2; W /= 2; }
      long e = (long)N * H * W * co[i]; if (e > maxE) maxE = e; } }
  __bf16* bufA = (__bf16*)(ws + off); off += (((size_t)maxE * 2 + 255) / 256) * 256;
  __bf16* bufB = (__bf16*)(ws + off);

  // ---- convert weights (transposed; layer 0 zero-padded to Cin=4) + input ----
  cc_cvt_w0T_pad4<<<3, 256, 0, stream>>>((const float*)d_in[1], wbf + woff[0]);
  for (int i = 1; i < 18; ++i) {
    long n = (long)welems[i];
    int blocks = (int)((n + 255) / 256); if (blocks > 4096) blocks = 4096;
    cc_cvt_wT<<<blocks, 256, 0, stream>>>((const float*)d_in[1 + i], wbf + woff[i],
                                          9 * ci[i], co[i], n);
  }
  cc_cvt_in_pad4<<<4096, 256, 0, stream>>>((const float*)d_in[0], bufA, (long)N * 768 * 256);

  // ---- layer loop ----
  __bf16* cur = bufA; __bf16* nxt = bufB;
  int H = 768, W = 256, C = 4;
  for (int i = 0; i < 18; ++i) {
    const int Ho = (st[i] == 2) ? H / 2 : H;
    const int Wo = (st[i] == 2) ? W / 2 : W;
    const long Mtot = (long)N * Ho * Wo;
    dim3 grid((unsigned)((Mtot + BM - 1) / BM), (unsigned)((co[i] + BN - 1) / BN));
    cc_conv3x3_wmma_bf16<<<grid, 256, 0, stream>>>(
        cur, wbf + woff[i], nxt,
        H, W, C, __builtin_ctz(C), Ho, Wo, __builtin_ctz(Wo), co[i],
        st[i], (i > 0) ? 1 : 0, Mtot);

    if (i >= 6) {
      const int j = i - 6;
      const long total = Mtot * co[i];
      const int cMask = co[i] - 1;
      long rb = (total + 255) / 256; if (rb > 2048) rb = 2048;
      rb = (rb + 1) & ~1L; if (rb < 2) rb = 2;         // even grid: stride % 512 == 0
      cc_zero_f32<<<(2 * co[i] + 255) / 256, 256, 0, stream>>>(stats, 2 * co[i]);
      cc_bn_reduce<<<(unsigned)rb, 256, 0, stream>>>(nxt, stats, total, co[i], cMask);
      cc_bn_apply<<<(unsigned)rb, 256, 0, stream>>>(nxt, stats,
          (const float*)d_in[19 + j], (const float*)d_in[31 + j],
          total, co[i], cMask, 1.0f / (float)Mtot);
    }
    { __bf16* t = cur; cur = nxt; nxt = t; }
    H = Ho; W = Wo; C = co[i];
  }

  // ---- head: avg pool over 12x4 -> [32,512], duplicated ----
  const int P = H * W;                 // 48
  dim3 pg((unsigned)N, (unsigned)((C + 255) / 256));
  cc_avgpool_head<<<pg, 256, 0, stream>>>(cur, (float*)d_out, P, C,
                                          1.0f / (float)P, (long)N * C);
}
